// DualStreamSAGE_27917287424478
// MI455X (gfx1250) — matrix operand order repaired and checked by reading, not compiled, not run on hardware
//
#include <hip/hip_runtime.h>

#define NN 100000
#define NE 3200000
#define NG 16
#define HID 64

typedef __attribute__((ext_vector_type(16))) _Float16 v16h;
typedef __attribute__((ext_vector_type(8)))  float    v8f;

struct H8 { _Float16 h[8]; };            // 16-byte chunk
union FragH { v16h v; _Float16 h[16]; H8 q[2]; };
union FragF { v8f v; float f[8]; };

// Fire-and-forget float max atomic -> GLOBAL_ATOMIC_MAX_NUM_F32 (no return, STOREcnt path)
__device__ __forceinline__ void amax_f32(float* p, float v) {
  asm volatile("global_atomic_max_num_f32 %0, %1, off scope:SCOPE_DEV"
               :: "v"(p), "v"(v) : "memory");
}

__global__ void k_fill(float* __restrict__ p, size_t n, float val) {
  size_t i = (size_t)blockIdx.x * blockDim.x + threadIdx.x;
  size_t stride = (size_t)gridDim.x * blockDim.x;
  for (; i < n; i += stride) p[i] = val;
}

// Pre-swizzle a 64x64 f32 weight into WMMA B-fragment layout (f16):
// Wf[((kc*4+nt)*32 + lane)*16 + j] = W[(kc*32 + hi*8 + j + (j>=8 ? 8 : 0))*64 + nt*16 + mrow]
__global__ void __launch_bounds__(256) k_prepw(
    const float* __restrict__ W, _Float16* __restrict__ Wf) {
  int t = threadIdx.x;          // 0..255 = frag(3b) x lane(5b)
  int lane = t & 31;
  int frag = t >> 5;            // kc*4 + nt
  int kc = frag >> 2, nt = frag & 3;
  int hi = lane >> 4, mrow = lane & 15;
  int n  = nt * 16 + mrow;
  int kb = kc * 32 + hi * 8;
  for (int j = 0; j < 16; ++j)
    Wf[t * 16 + j] = (_Float16)W[(kb + j + (j >= 8 ? 8 : 0)) * HID + n];
}

// ---- layer 1: scatter-max of x[src] (2 channels) into agg1 ----
__global__ void __launch_bounds__(256) k_scatter1(
    const int* __restrict__ src, const int* __restrict__ dst,
    const float* __restrict__ x, float* __restrict__ agg1, int nE) {
  int e = blockIdx.x * 256 + threadIdx.x;
  if (e >= nE) return;
  int s = src[e], d = dst[e];
  amax_f32(agg1 + (size_t)d * 2 + 0, x[(size_t)s * 2 + 0]);
  amax_f32(agg1 + (size_t)d * 2 + 1, x[(size_t)s * 2 + 1]);
}

// ---- layers 2/3: scatter-max of f16 h[src] (64 channels, 1 wave per edge) ----
__global__ void __launch_bounds__(256) k_scatter64(
    const int* __restrict__ src, const int* __restrict__ dst,
    const _Float16* __restrict__ h, float* __restrict__ agg, int nE) {
  long long t = (long long)blockIdx.x * 256 + threadIdx.x;
  long long e = t >> 5;
  int c2 = (int)(t & 31);
  if (e >= nE) return;
  int s = 0, d = 0;
  if (c2 == 0) { s = src[e]; d = dst[e]; }
  s = __shfl(s, 0); d = __shfl(d, 0);
  union { unsigned u; _Float16 h2[2]; } pk;
  pk.u = *(const unsigned*)(h + (size_t)s * HID + c2 * 2);  // single b32 load
  float* ap = agg + (size_t)d * HID + c2 * 2;
  amax_f32(ap + 0, (float)pk.h2[0]);
  amax_f32(ap + 1, (float)pk.h2[1]);
}

// SAGE layer 1 (fan-in 2): h1 = relu(agg1@Wl + bl + x@Wr), thread per (node, channel)
__global__ void __launch_bounds__(256) k_node1(
    const float* __restrict__ agg1, const float* __restrict__ x,
    const float* __restrict__ Wl, const float* __restrict__ bl,
    const float* __restrict__ Wr, _Float16* __restrict__ h1) {
  int t = blockIdx.x * 256 + threadIdx.x;
  if (t >= NN * HID) return;
  int i = t >> 6, c = t & 63;
  float a0 = agg1[i * 2], a1 = agg1[i * 2 + 1];
  a0 = (a0 < -1e37f) ? 0.f : a0;
  a1 = (a1 < -1e37f) ? 0.f : a1;
  float v = a0 * Wl[c] + a1 * Wl[64 + c] + bl[c]
          + x[i * 2] * Wr[c] + x[i * 2 + 1] * Wr[64 + c];
  h1[t] = (_Float16)(v > 0.f ? v : 0.f);
}

// eco layer 1 (fan-in 2): t1 = relu([costs, budget[batch]]@W1 + b1)
__global__ void __launch_bounds__(256) k_eco1(
    const float* __restrict__ costs, const float* __restrict__ budget,
    const int* __restrict__ bid, const float* __restrict__ W1,
    const float* __restrict__ b1, _Float16* __restrict__ t1) {
  int t = blockIdx.x * 256 + threadIdx.x;
  if (t >= NN * HID) return;
  int i = t >> 6, c = t & 63;
  float in0 = costs[i];
  float in1 = budget[bid[i]];
  float v = in0 * W1[c] + in1 * W1[64 + c] + b1[c];
  t1[t] = (_Float16)(v > 0.f ? v : 0.f);
}

// ---- WMMA GEMM: OUT[N,64] = relu(A1@W1 (+ A2@W2) + bias), 1 wave per 16-row tile.
// A1 either f16 row-major, or raw f32 agg buffer (cleaned -inf->0 and cvt in regs).
// Weights come pre-swizzled in fragment layout (k_prepw): 2 x b128 per fragment.
template <bool A1F32, bool DUAL, bool OUT16>
__global__ void __launch_bounds__(256) gemm64(
    const _Float16* __restrict__ A1h, const float* __restrict__ A1f,
    const _Float16* __restrict__ A2,
    const _Float16* __restrict__ Wf1, const _Float16* __restrict__ Wf2,
    const float* __restrict__ bias,
    _Float16* __restrict__ o16, float* __restrict__ o32, int nTiles) {
  int wave = threadIdx.x >> 5;
  int lane = threadIdx.x & 31;
  int tile = blockIdx.x * 8 + wave;
  if (tile >= nTiles) return;   // uniform per wave

  int mrow = lane & 15;
  int hi   = lane >> 4;         // 0: lanes 0-15, 1: lanes 16-31
  int base = tile * 16;

  // A fragments (16-bit A 16x32 layout): lane holds row mrow, K = hi*8 + [0..7] and +16
  FragH a1[2], a2[2];
  for (int kc = 0; kc < 2; ++kc) {
    int kb = kc * 32 + hi * 8;
    if (A1F32) {
      const float* pa = A1f + (size_t)(base + mrow) * HID + kb;
      v8f u0 = *(const v8f*)pa;         // 32B vector load
      v8f u1 = *(const v8f*)(pa + 16);
      for (int j = 0; j < 8; ++j) {
        float v0 = u0[j]; v0 = (v0 < -1e37f) ? 0.f : v0;
        float v1 = u1[j]; v1 = (v1 < -1e37f) ? 0.f : v1;
        a1[kc].h[j]     = (_Float16)v0;
        a1[kc].h[8 + j] = (_Float16)v1;
      }
    } else {
      const _Float16* pa = A1h + (size_t)(base + mrow) * HID + kb;
      a1[kc].q[0] = *(const H8*)pa;
      a1[kc].q[1] = *(const H8*)(pa + 16);
    }
    if (DUAL) {
      const _Float16* pb = A2 + (size_t)(base + mrow) * HID + kb;
      a2[kc].q[0] = *(const H8*)pb;
      a2[kc].q[1] = *(const H8*)(pb + 16);
    }
  }

  const H8* w1 = (const H8*)Wf1;
  const H8* w2 = (const H8*)Wf2;
  for (int nt = 0; nt < 4; ++nt) {
    int n = nt * 16 + mrow;
    // B fragments from pre-swizzled layout: 2 x 16B contiguous per lane, coalesced
    FragH b1[2], b2[2];
    for (int kc = 0; kc < 2; ++kc) {
      int idx = ((kc * 4 + nt) * 32 + lane) * 2;
      b1[kc].q[0] = w1[idx];
      b1[kc].q[1] = w1[idx + 1];
      if (DUAL) {
        b2[kc].q[0] = w2[idx];
        b2[kc].q[1] = w2[idx + 1];
      }
    }
    FragF acc;
    for (int r = 0; r < 8; ++r) acc.f[r] = 0.f;
    acc.v = __builtin_amdgcn_wmma_f32_16x16x32_f16(false, a1[0].v, false, b1[0].v,
                                                   (short)0, acc.v, false, false);
    acc.v = __builtin_amdgcn_wmma_f32_16x16x32_f16(false, a1[1].v, false, b1[1].v,
                                                   (short)0, acc.v, false, false);
    if (DUAL) {
      acc.v = __builtin_amdgcn_wmma_f32_16x16x32_f16(false, a2[0].v, false, b2[0].v,
                                                     (short)0, acc.v, false, false);
      acc.v = __builtin_amdgcn_wmma_f32_16x16x32_f16(false, a2[1].v, false, b2[1].v,
                                                     (short)0, acc.v, false, false);
    }
    float bn = bias[n];
    for (int r = 0; r < 8; ++r) {
      int m = base + r + hi * 8;   // C/D layout: VGPR r -> row r (+8 for lanes 16-31)
      float val = acc.f[r] + bn;
      val = val > 0.f ? val : 0.f;
      if (OUT16) o16[(size_t)m * HID + n] = (_Float16)val;
      else       o32[(size_t)m * HID + n] = val;
    }
  }
}

// gate + combine + graph max-pool (1 wave per node)
__global__ void __launch_bounds__(256) k_gate(
    const float* __restrict__ hT, const float* __restrict__ hE,
    const float* __restrict__ W1, const float* __restrict__ b1,
    const float* __restrict__ W2, const float* __restrict__ b2,
    const int* __restrict__ bid, float* __restrict__ alphaOut,
    float* __restrict__ hF, float* __restrict__ gemb) {
  int wave = threadIdx.x >> 5, lane = threadIdx.x & 31;
  int i = blockIdx.x * 8 + wave;
  if (i >= NN) return;
  const float* ht = hT + (size_t)i * HID;
  const float* he = hE + (size_t)i * HID;
  float g = b1[lane];
  for (int k = 0; k < HID; ++k) g += ht[k] * W1[k * 32 + lane];
  for (int k = 0; k < HID; ++k) g += he[k] * W1[(HID + k) * 32 + lane];
  g = tanhf(g);
  float p = g * W2[lane];
  for (int off = 16; off > 0; off >>= 1) p += __shfl_down(p, off);
  p = __shfl(p, 0);
  float alpha = 1.f / (1.f + expf(-(p + b2[0])));
  if (lane == 0) alphaOut[i] = alpha;
  int b = bid[i];
  for (int c = lane; c < HID; c += 32) {
    float v = alpha * ht[c] + (1.f - alpha) * he[c];
    hF[(size_t)i * HID + c] = v;
    amax_f32(gemb + (size_t)b * HID + c, v);
  }
}

// actor head (1 wave per node): logits = relu(hF@W1+b1)@W2 + b2
__global__ void __launch_bounds__(256) k_actor(
    const float* __restrict__ hF, const float* __restrict__ W1,
    const float* __restrict__ b1, const float* __restrict__ W2,
    const float* __restrict__ b2, float* __restrict__ out) {
  int wave = threadIdx.x >> 5, lane = threadIdx.x & 31;
  int i = blockIdx.x * 8 + wave;
  if (i >= NN) return;
  const float* hf = hF + (size_t)i * HID;
  float t0 = b1[lane], t1 = b1[lane + 32];
  for (int k = 0; k < HID; ++k) {
    float hk = hf[k];
    t0 += hk * W1[k * HID + lane];
    t1 += hk * W1[k * HID + lane + 32];
  }
  t0 = t0 > 0.f ? t0 : 0.f;
  t1 = t1 > 0.f ? t1 : 0.f;
  float p = t0 * W2[lane] + t1 * W2[lane + 32];
  for (int off = 16; off > 0; off >>= 1) p += __shfl_down(p, off);
  if (lane == 0) out[i] = p + b2[0];
}

// critic head over pooled graph embeddings (16 graphs)
__global__ void __launch_bounds__(64) k_critic(
    const float* __restrict__ gemb, const float* __restrict__ W1,
    const float* __restrict__ b1, const float* __restrict__ W2,
    const float* __restrict__ b2, float* __restrict__ out) {
  __shared__ float emb[HID];
  __shared__ float red[HID];
  int g = blockIdx.x, c = threadIdx.x;
  float e = gemb[(size_t)g * HID + c];
  emb[c] = (e < -1e37f) ? 0.f : e;
  __syncthreads();
  float t = b1[c];
  for (int k = 0; k < HID; ++k) t += emb[k] * W1[k * HID + c];
  t = t > 0.f ? t : 0.f;
  red[c] = t * W2[c];
  __syncthreads();
  if (c == 0) {
    float s = 0.f;
    for (int k = 0; k < HID; ++k) s += red[k];
    out[g] = s + b2[0];
  }
}

extern "C" void kernel_launch(void* const* d_in, const int* in_sizes, int n_in,
                              void* d_out, int out_size, void* d_ws, size_t ws_size,
                              hipStream_t stream) {
  const float* x      = (const float*)d_in[0];
  const int*   src    = (const int*)d_in[1];
  const int*   dst    = src + NE;
  const float* costs  = (const float*)d_in[2];
  const float* budget = (const float*)d_in[3];
  const int*   bid    = (const int*)d_in[4];
  const float* s1Wl = (const float*)d_in[5];
  const float* s1bl = (const float*)d_in[6];
  const float* s1Wr = (const float*)d_in[7];
  const float* s2Wl = (const float*)d_in[8];
  const float* s2bl = (const float*)d_in[9];
  const float* s2Wr = (const float*)d_in[10];
  const float* s3Wl = (const float*)d_in[11];
  const float* s3bl = (const float*)d_in[12];
  const float* s3Wr = (const float*)d_in[13];
  const float* eW1  = (const float*)d_in[14];
  const float* eb1  = (const float*)d_in[15];
  const float* eW2  = (const float*)d_in[16];
  const float* eb2  = (const float*)d_in[17];
  const float* gW1  = (const float*)d_in[18];
  const float* gb1  = (const float*)d_in[19];
  const float* gW2  = (const float*)d_in[20];
  const float* gb2  = (const float*)d_in[21];
  const float* aW1  = (const float*)d_in[22];
  const float* ab1  = (const float*)d_in[23];
  const float* aW2  = (const float*)d_in[24];
  const float* ab2  = (const float*)d_in[25];
  const float* cW1  = (const float*)d_in[26];
  const float* cb1  = (const float*)d_in[27];
  const float* cW2  = (const float*)d_in[28];
  const float* cb2  = (const float*)d_in[29];

  float* out_logits = (float*)d_out;
  float* out_value  = (float*)d_out + NN;
  float* out_alpha  = (float*)d_out + NN + NG;

  char* ws = (char*)d_ws;
  auto alloc = [&](size_t bytes) {
    char* p = ws;
    ws += (bytes + 255) & ~(size_t)255;
    return p;
  };
  float*    agg64 = (float*)   alloc((size_t)NN * HID * 4);
  _Float16* h1    = (_Float16*)alloc((size_t)NN * HID * 2);
  _Float16* h2    = (_Float16*)alloc((size_t)NN * HID * 2);
  _Float16* t1    = (_Float16*)alloc((size_t)NN * HID * 2);
  float*    hEco  = (float*)   alloc((size_t)NN * HID * 4);
  float*    hTopo = (float*)   alloc((size_t)NN * HID * 4);
  float*    hFin  = (float*)   alloc((size_t)NN * HID * 4);
  float*    agg1  = (float*)   alloc((size_t)NN * 2 * 4);
  float*    gemb  = (float*)   alloc((size_t)NG * HID * 4);
  _Float16* wfE2  = (_Float16*)alloc(HID * HID * 2);  // pre-swizzled weights
  _Float16* wfS2l = (_Float16*)alloc(HID * HID * 2);
  _Float16* wfS2r = (_Float16*)alloc(HID * HID * 2);
  _Float16* wfS3l = (_Float16*)alloc(HID * HID * 2);
  _Float16* wfS3r = (_Float16*)alloc(HID * HID * 2);

  const float NEG = -__builtin_inff();
  const int nodeC = (NN * HID) / 256;             // 25000
  const int tiles = NN / 16;                      // 6250
  const int gemmB = (tiles + 7) / 8;              // 782
  const int edgeB = (NE + 255) / 256;             // 12500
  const int edgeW = (int)(((long long)NE * 32) / 256); // 400000
  const int nodeW = (NN + 7) / 8;                 // 12500

  // init + weight pre-swizzle
  k_fill<<<1024, 256, 0, stream>>>(agg1, (size_t)NN * 2, NEG);
  k_fill<<<16, 256, 0, stream>>>(gemb, (size_t)NG * HID, NEG);
  k_prepw<<<1, 256, 0, stream>>>(eW2,  wfE2);
  k_prepw<<<1, 256, 0, stream>>>(s2Wl, wfS2l);
  k_prepw<<<1, 256, 0, stream>>>(s2Wr, wfS2r);
  k_prepw<<<1, 256, 0, stream>>>(s3Wl, wfS3l);
  k_prepw<<<1, 256, 0, stream>>>(s3Wr, wfS3r);

  // SAGE layer 1 (fan-in 2, VALU)
  k_scatter1<<<edgeB, 256, 0, stream>>>(src, dst, x, agg1, NE);
  k_node1<<<nodeC, 256, 0, stream>>>(agg1, x, s1Wl, s1bl, s1Wr, h1);

  // eco path
  k_eco1<<<nodeC, 256, 0, stream>>>(costs, budget, bid, eW1, eb1, t1);
  gemm64<false, false, false><<<gemmB, 256, 0, stream>>>(
      t1, nullptr, nullptr, wfE2, nullptr, eb2, nullptr, hEco, tiles);

  // SAGE layer 2 (WMMA; A1 = raw f32 agg, cleaned in-kernel)
  k_fill<<<2048, 256, 0, stream>>>(agg64, (size_t)NN * HID, NEG);
  k_scatter64<<<edgeW, 256, 0, stream>>>(src, dst, h1, agg64, NE);
  gemm64<true, true, true><<<gemmB, 256, 0, stream>>>(
      nullptr, agg64, h1, wfS2l, wfS2r, s2bl, h2, nullptr, tiles);

  // SAGE layer 3 (WMMA)
  k_fill<<<2048, 256, 0, stream>>>(agg64, (size_t)NN * HID, NEG);
  k_scatter64<<<edgeW, 256, 0, stream>>>(src, dst, h2, agg64, NE);
  gemm64<true, true, false><<<gemmB, 256, 0, stream>>>(
      nullptr, agg64, h2, wfS3l, wfS3r, s3bl, nullptr, hTopo, tiles);

  // gate + combine + pool, heads
  k_gate<<<nodeW, 256, 0, stream>>>(hTopo, hEco, gW1, gb1, gW2, gb2, bid,
                                    out_alpha, hFin, gemb);
  k_actor<<<nodeW, 256, 0, stream>>>(hFin, aW1, ab1, aW2, ab2, out_logits);
  k_critic<<<NG, 64, 0, stream>>>(gemb, cW1, cb1, cW2, cb2, out_value);
}
